// HyperGen_32375463477390
// MI455X (gfx1250) — matrix operand order repaired and checked by reading, not compiled, not run on hardware
//
#include <hip/hip_runtime.h>
#include <stdint.h>

// Problem constants (from reference)
#define COUTC 64
#define CINC  64
#define HH    256
#define WW    256
#define SCN   128
#define PTOT  36864                  // COUT*CIN*3*3
#define NPATCH 512                   // B * FH * FW
#define KDIM  576                    // CIN*9
#define XSTR  20                     // padded LDS row stride for 18-wide window

typedef __attribute__((ext_vector_type(16))) __bf16          v16bf;
typedef __attribute__((ext_vector_type(16))) unsigned short  v16u;
typedef __attribute__((ext_vector_type(8)))  float           v8f;
typedef __attribute__((ext_vector_type(4)))  unsigned int    u32x4;
typedef __attribute__((ext_vector_type(8)))  int             i32x8;
typedef __attribute__((ext_vector_type(4)))  int             i32x4;

#if __has_builtin(__builtin_amdgcn_tensor_load_to_lds) && \
    __has_builtin(__builtin_amdgcn_s_wait_tensorcnt)
#define USE_TDM 1
#else
#define USE_TDM 0
#endif

__device__ __forceinline__ unsigned short f2bf(float f) {
  union { float f; uint32_t u; } a; a.f = f;
  uint32_t u = a.u;
  uint32_t r = (u + 0x7FFFu + ((u >> 16) & 1u)) >> 16;  // round-to-nearest-even
  return (unsigned short)r;
}

__device__ __forceinline__ v8f wmma_bf16(v16u a, v16u b, v8f c) {
  union { v16u u; v16bf b; } A, B;
  A.u = a; B.u = b;
  return __builtin_amdgcn_wmma_f32_16x16x32_bf16(
      /*neg_a=*/false, A.b, /*neg_b=*/false, B.b,
      /*c_mod=*/(short)0, c, /*reuse_a=*/false, /*reuse_b=*/false);
}

// ---------------- kernel 0a: s2w f32 -> bf16 (same layout [P][SC]) ----------
__global__ void k_cvt_s2w(const float* __restrict__ s2w,
                          unsigned short* __restrict__ o, int n) {
  int i = blockIdx.x * 256 + threadIdx.x;
  if (i < n) o[i] = f2bf(s2w[i]);
}

// ---------------- kernel 0b: s [B][SC][FH][FW] -> bf16 [patch][SC] ----------
__global__ void k_cvt_s(const float* __restrict__ s,
                        unsigned short* __restrict__ o) {
  int i = blockIdx.x * 256 + threadIdx.x;        // i = patch*128 + c
  if (i >= NPATCH * SCN) return;
  int c     = i & (SCN - 1);
  int patch = i >> 7;
  int b     = patch >> 8;
  int fg    = patch & 255;
  o[i] = f2bf(s[(size_t)(b * SCN + c) * 256 + fg]);
}

// ---------------- kernel 1: Wdyn = s2w @ s  (bf16 WMMA GEMM) ----------------
// A = s2w_bf [36864][128], B = s_t [512][128] (row = patch, K contiguous)
// D -> Wd bf16 [patch][36864]; p = cout*576 + cin*9 + i*3 + j
__global__ void k_gen_w(const unsigned short* __restrict__ A,
                        const unsigned short* __restrict__ Bm,
                        unsigned short* __restrict__ Wd) {
  const int lane = threadIdx.x & 31;
  const int wave = threadIdx.x >> 5;      // 0..7
  const int m0   = blockIdx.x * 16;       // M tile (rows of P)
  const int hi   = lane >> 4;             // half-wave select
  const int lm   = lane & 15;

  v8f acc[4];
  #pragma unroll
  for (int t = 0; t < 4; ++t) acc[t] = (v8f){0.f,0.f,0.f,0.f,0.f,0.f,0.f,0.f};

  for (int k0 = 0; k0 < SCN; k0 += 32) {
    // A fragment: 16-bit A 16x32 layout. lanes0-15: V0-3 K0..7, V4-7 K16..23;
    // lanes16-31: +8.  (Two contiguous 16B chunks -> b128 loads.)
    const unsigned short* arow = A + (size_t)(m0 + lm) * SCN + k0 + hi * 8;
    v16u af;
    #pragma unroll
    for (int v = 0; v < 8; ++v) {
      int K = (v < 4 ? 0 : 16) + 2 * (v & 3);
      uint32_t d = *(const uint32_t*)(arow + K);
      af[2*v]   = (unsigned short)d;
      af[2*v+1] = (unsigned short)(d >> 16);
    }
    #pragma unroll
    for (int t = 0; t < 4; ++t) {
      int ntile = wave + t * 8;           // 32 N-tiles of patches
      const unsigned short* brow =
          Bm + (size_t)(ntile * 16 + lm) * SCN + k0 + hi * 8;
      v16u bf;
      #pragma unroll
      for (int v = 0; v < 8; ++v) {       // B: same K layout as A (K0..7,16..23; lanes16-31 +8)
        uint32_t d = *(const uint32_t*)(brow + (v < 4 ? 0 : 16) + 2 * (v & 3));
        bf[2*v]   = (unsigned short)d;
        bf[2*v+1] = (unsigned short)(d >> 16);
      }
      acc[t] = wmma_bf16(af, bf, acc[t]);
    }
  }
  // C/D layout: VGPR r -> M = r + hi*8, N = lm
  #pragma unroll
  for (int t = 0; t < 4; ++t) {
    int n = (wave + t * 8) * 16 + lm;     // patch index
    #pragma unroll
    for (int r = 0; r < 8; ++r) {
      int m = m0 + r + hi * 8;            // p index
      Wd[(size_t)n * PTOT + m] = f2bf(acc[t][r]);
    }
  }
}

// ---------------- kernel 2: per-patch dynamic 3x3 conv (implicit GEMM) -----
// One block = one patch (8 waves). LDS holds:
//   sW: the 72KB bf16 weight tile, staged by the Tensor Data Mover (TDM)
//   sX: the 18x18x64 reflect-padded bf16 window
// TDM DMA overlaps with the window staging done by the VALU/VMEM pipes.
__global__ void k_conv(const float* __restrict__ x,
                       const unsigned short* __restrict__ Wd,
                       float* __restrict__ out) {
  __shared__ unsigned short sW[PTOT];               // 73728 bytes
  __shared__ unsigned short sX[CINC * 18 * XSTR];   // 46080 bytes

  const int patch = blockIdx.x;
  const int b = patch >> 8;
  const int f = (patch >> 4) & 15;
  const int g = patch & 15;
  const int tid = threadIdx.x;

#if USE_TDM
  // One wave issues a 1-D TDM descriptor: 9216 x 8B = 72KB, global -> LDS.
  if (tid < 32) {
    uint64_t ga   = (uint64_t)(uintptr_t)(Wd + (size_t)patch * PTOT);
    unsigned lds0 = (unsigned)(uintptr_t)(void*)&sW[0];  // flat->LDS = addr[31:0]
    u32x4 g0;
    g0[0] = 1u;                                   // count=1, user descriptor
    g0[1] = lds0;                                 // lds_addr (bytes)
    g0[2] = (unsigned)ga;                         // global_addr[31:0]
    g0[3] = ((unsigned)(ga >> 32) & 0x01FFFFFFu)  // global_addr[56:32]
            | (2u << 30);                         // type=2 ("image")
    i32x8 g1;
    g1[0] = 3 << 16;                              // data_size=8B; wg_mask=0
    g1[1] = (int)(9216u << 16);                   // tensor_dim0 = 9216 (low16)
    g1[2] = (int)(1u << 16);                      // tensor_dim1 = 1
    g1[3] = (int)(9216u << 16);                   // tile_dim0 = 9216
    g1[4] = 0;                                    // tile_dim1/2 unused
    g1[5] = 9216;                                 // tensor_dim0_stride
    g1[6] = 0;
    g1[7] = 0;
    i32x4 gz = {0, 0, 0, 0};
#if __clang_major__ >= 23
    i32x8 gz8 = {0, 0, 0, 0, 0, 0, 0, 0};
    __builtin_amdgcn_tensor_load_to_lds(g0, g1, gz, gz, gz8, 0);
#else
    __builtin_amdgcn_tensor_load_to_lds(g0, g1, gz, gz, 0);
#endif
  }
#endif

  // Stage window with reflect padding (pad=1), f32 -> bf16 (overlaps TDM DMA)
  const int y0 = f * 16 - 1, x0 = g * 16 - 1;
  for (int idx = tid; idx < CINC * 18 * 18; idx += 256) {
    int c   = idx / 324;
    int rem = idx - c * 324;
    int yy  = rem / 18;
    int xx  = rem - yy * 18;
    int ys = y0 + yy; ys = ys < 0 ? -ys : (ys > 255 ? 510 - ys : ys);
    int xs = x0 + xx; xs = xs < 0 ? -xs : (xs > 255 ? 510 - xs : xs);
    float v = x[((size_t)(b * CINC + c) * HH + ys) * WW + xs];
    sX[(c * 18 + yy) * XSTR + xx] = f2bf(v);
  }

#if USE_TDM
  if (tid < 32) __builtin_amdgcn_s_wait_tensorcnt((short)0);
#else
  {
    const uint4* src = (const uint4*)(Wd + (size_t)patch * PTOT);
    uint4* dst = (uint4*)sW;
    for (int i = tid; i < PTOT / 8; i += 256) dst[i] = src[i];
  }
#endif
  __syncthreads();

  const int lane  = tid & 31;
  const int wave  = tid >> 5;
  const int hi    = lane >> 4;
  const int lm    = lane & 15;
  const int mtile = wave & 3;             // cout tile 0..3
  const int nbase = (wave >> 2) * 8;      // pixel-row tiles: 0..7 or 8..15

  const unsigned short* sWp = sW + (size_t)(mtile * 16 + lm) * KDIM + hi * 8;

  v8f acc[8];
  #pragma unroll
  for (int t = 0; t < 8; ++t) acc[t] = (v8f){0.f,0.f,0.f,0.f,0.f,0.f,0.f,0.f};

  for (int k0 = 0; k0 < KDIM; k0 += 32) {
    // A fragment from LDS: two 16B-aligned chunks -> ds_load_b128 pair
    v16u af;
    #pragma unroll
    for (int v = 0; v < 8; ++v) {
      int K = (v < 4 ? 0 : 16) + 2 * (v & 3);
      uint32_t d = *(const uint32_t*)(sWp + k0 + K);
      af[2*v]   = (unsigned short)d;
      af[2*v+1] = (unsigned short)(d >> 16);
    }
    // Per-K LDS base offsets, shared by all 8 N-tiles of this wave
    int boff[16];
    #pragma unroll
    for (int v = 0; v < 8; ++v) {
      #pragma unroll
      for (int h = 0; h < 2; ++h) {
        int kk = k0 + hi * 8 + (v < 4 ? 0 : 16) + 2 * (v & 3) + h;    // kdim index = c*9 + i*3 + j
        int c  = kk / 9;
        int r9 = kk - c * 9;
        int i  = r9 / 3;
        int j  = r9 - i * 3;
        boff[2*v+h] = (c * 18 + i) * XSTR + j;
      }
    }
    #pragma unroll
    for (int t = 0; t < 8; ++t) {
      int poff = (nbase + t) * XSTR + lm;     // + p*XSTR + q
      v16u bf;
      #pragma unroll
      for (int e = 0; e < 16; ++e) bf[e] = sX[boff[e] + poff];
      acc[t] = wmma_bf16(af, bf, acc[t]);
    }
  }

  // Store f32 output: out[b][co][f*16+p][g*16+q]
  #pragma unroll
  for (int t = 0; t < 8; ++t) {
    int p  = nbase + t;
    int yy = f * 16 + p, xx = g * 16 + lm;
    #pragma unroll
    for (int r = 0; r < 8; ++r) {
      int co = mtile * 16 + r + hi * 8;
      out[((size_t)(b * COUTC + co) * HH + yy) * WW + xx] = acc[t][r];
    }
  }
}

// ---------------------------------------------------------------------------
extern "C" void kernel_launch(void* const* d_in, const int* in_sizes, int n_in,
                              void* d_out, int out_size, void* d_ws, size_t ws_size,
                              hipStream_t stream) {
  const float* x   = (const float*)d_in[0];
  const float* s   = (const float*)d_in[1];
  const float* s2w = (const float*)d_in[2];
  float* out = (float*)d_out;

  // Workspace layout (bf16 = ushort):
  //   Wd    : [512][36864]      = 37,748,736 B
  //   s2w_bf: [36864][128]      =  9,437,184 B
  //   s_t   : [512][128]        =    131,072 B
  unsigned short* wd   = (unsigned short*)d_ws;
  unsigned short* a_bf = wd + (size_t)NPATCH * PTOT;
  unsigned short* b_bf = a_bf + (size_t)PTOT * SCN;

  k_cvt_s2w<<<(PTOT * SCN + 255) / 256, 256, 0, stream>>>(s2w, a_bf, PTOT * SCN);
  k_cvt_s<<<(NPATCH * SCN + 255) / 256, 256, 0, stream>>>(s, b_bf);
  k_gen_w<<<PTOT / 16, 256, 0, stream>>>(a_bf, b_bf, wd);
  k_conv<<<NPATCH, 256, 0, stream>>>(x, wd, out);
}
